// DynamicGatedAttention_15229954031654
// MI455X (gfx1250) — compile-verified, matrix operand order checked by reading, not run on hardware
//
#include <hip/hip_runtime.h>

typedef __attribute__((ext_vector_type(16))) _Float16 v16h;
typedef __attribute__((ext_vector_type(8)))  float    v8f;

#define EPSV 1e-5f
#define PI_F 3.14159265358979323846f

union V16 { v16h v; _Float16 h[16]; };

__device__ __forceinline__ v8f wmma16(v16h a, v16h b, v8f c) {
  return __builtin_amdgcn_wmma_f32_16x16x32_f16(false, a, false, b, (short)0, c,
                                                false, false);
}

__device__ __forceinline__ v8f vzero8() {
  v8f z = {0.f, 0.f, 0.f, 0.f, 0.f, 0.f, 0.f, 0.f};
  return z;
}

// Gather a 16-bit WMMA B fragment (column nn) directly from a row-major f32
// weight W[k*ldn + n]. Element e holds K = kbase+8*hi+e (e<8) and
// K = kbase+16+8*hi+(e-8) (e>=8), zero-padded outside [0,kmax)x[0,nmax).
__device__ __forceinline__ v16h gathW(const float* __restrict__ W, int ldn, int nn,
                                      int kbase, int hi, int kmax, int nmax) {
  V16 u;
#pragma unroll
  for (int e = 0; e < 8; ++e) {
    int k0 = kbase + 8 * hi + e, k1 = kbase + 16 + 8 * hi + e;
    u.h[e]     = (k0 < kmax && nn < nmax) ? (_Float16)W[k0 * ldn + nn] : (_Float16)0.f;
    u.h[8 + e] = (k1 < kmax && nn < nmax) ? (_Float16)W[k1 * ldn + nn] : (_Float16)0.f;
  }
  return u.v;
}

__device__ __forceinline__ _Float16 hsel(int hi, float a, float b) {
  return (_Float16)(hi ? b : a);
}

// ---------------------------------------------------------------------------
// Projection kernel: one launch per matrix (q/k/v). 8 weight fragments held
// in registers; BN folded into per-lane scale/shift registers.
// ---------------------------------------------------------------------------
__global__ void proj_kernel(const float* __restrict__ feat, const float* __restrict__ W,
                            const float* __restrict__ bias, const float* __restrict__ bn,
                            float* __restrict__ dst, int Npts, int hasBN) {
  int tid = threadIdx.x;
  int w = tid >> 5, lane = tid & 31, r = lane & 15, hi = lane >> 4;

  V16 bf[4][2];
  float sc[4], sh[4];
#pragma unroll
  for (int t4 = 0; t4 < 4; ++t4) {
    int ch = r + 16 * t4;
    bf[t4][0].v = gathW(W, 64, ch, 0, hi, 64, 64);
    bf[t4][1].v = gathW(W, 64, ch, 32, hi, 64, 64);
    if (hasBN) {
      float s = bn[ch] * rsqrtf(bn[192 + ch] + EPSV);
      sc[t4] = s;
      sh[t4] = (bias[ch] - bn[128 + ch]) * s + bn[64 + ch];
    } else {
      sc[t4] = 1.f;
      sh[t4] = bias[ch];
    }
  }

  int tiles = (Npts + 15) / 16;
  for (int tile = blockIdx.x * 8 + w; tile < tiles; tile += gridDim.x * 8) {
    int row = tile * 16 + r;
    int rowc = (row < Npts) ? row : (Npts - 1);
    const float* fr = feat + (size_t)rowc * 64;
    V16 a0, a1;
#pragma unroll
    for (int e = 0; e < 8; ++e) {
      a0.h[e]     = (_Float16)fr[8 * hi + e];
      a0.h[8 + e] = (_Float16)fr[16 + 8 * hi + e];
      a1.h[e]     = (_Float16)fr[32 + 8 * hi + e];
      a1.h[8 + e] = (_Float16)fr[48 + 8 * hi + e];
    }
    bool fullTile = (tile * 16 + 16 <= Npts);
#pragma unroll
    for (int t4 = 0; t4 < 4; ++t4) {
      int ch = r + 16 * t4;
      v8f acc = vzero8();
      acc = wmma16(a0.v, bf[t4][0].v, acc);
      acc = wmma16(a1.v, bf[t4][1].v, acc);
      if (fullTile) {
        float* p = dst + ((size_t)(tile * 16 + 8 * hi)) * 64 + ch;
#pragma unroll
        for (int i = 0; i < 8; ++i) {
          float val = acc[i] * sc[t4] + sh[t4];
          if (hasBN) val = fmaxf(val, 0.f);
          p[(size_t)i * 64] = val;
        }
      } else {
#pragma unroll
        for (int i = 0; i < 8; ++i) {
          float val = acc[i] * sc[t4] + sh[t4];
          if (hasBN) val = fmaxf(val, 0.f);
          int m = tile * 16 + i + 8 * hi;
          m = (m < Npts) ? m : (Npts - 1);
          dst[(size_t)m * 64 + ch] = val;
        }
      }
    }
  }
}

// ---------------------------------------------------------------------------
// Segment max of q (q >= 0 post-ReLU, so uint-bit atomicMax == float max).
// ---------------------------------------------------------------------------
__global__ void seg_init(float* segm) {
  int t = threadIdx.x;
  if (t < 256) segm[t] = 0.f;
}

__global__ void seg_max(const float* __restrict__ q, const int* __restrict__ offs,
                        unsigned int* __restrict__ segm, int Npts) {
  int o0 = offs[0], o1 = offs[1], o2 = offs[2], o3 = offs[3];
  long total = (long)Npts * 64;
  long stride = (long)gridDim.x * blockDim.x;
  for (long i = (long)blockIdx.x * blockDim.x + threadIdx.x; i < total; i += stride) {
    int row = (int)(i >> 6), c = (int)(i & 63);
    int s = (row >= o0) + (row >= o1) + (row >= o2) + (row >= o3);
    atomicMax(&segm[s * 64 + c], __float_as_uint(q[i]));
  }
}

// ---------------------------------------------------------------------------
// Gate MLP: gate_in = [q, max_k q[idx], seg_max] (192) -> 64 (bn,relu) -> 8,
// sigmoid. One wave per point.
// ---------------------------------------------------------------------------
__global__ void gate_kernel(const float* __restrict__ qws, const float* __restrict__ segm,
                            const int* __restrict__ refidx, const int* __restrict__ offs,
                            const float* __restrict__ W1, const float* __restrict__ b1,
                            const float* __restrict__ bn1,
                            const float* __restrict__ W2, const float* __restrict__ b2,
                            float* __restrict__ gate, int Npts, int tiles) {
  __shared__ float gin[8][192];
  __shared__ float hbuf[8][64];
  int tid = threadIdx.x, w = tid >> 5, lane = tid & 31;
  int o0 = offs[0], o1 = offs[1], o2 = offs[2], o3 = offs[3];
  for (int t = blockIdx.x; t < tiles; t += gridDim.x) {
    int n = t * 8 + w;
    bool act = (n < Npts);
    if (act) {
      int c0 = 2 * lane;
      const float* qr = qws + (size_t)n * 64;
      float nm0 = -1e30f, nm1 = -1e30f;
      for (int j = 0; j < 16; ++j) {
        int im = refidx[n * 16 + j]; im = (im < 0) ? 0 : im;
        float2 qq = *(const float2*)(qws + (size_t)im * 64 + c0);
        nm0 = fmaxf(nm0, qq.x); nm1 = fmaxf(nm1, qq.y);
      }
      int s = (n >= o0) + (n >= o1) + (n >= o2) + (n >= o3);
      float2 qv = *(const float2*)(qr + c0);
      float2 gv = *(const float2*)(segm + s * 64 + c0);
      gin[w][c0] = qv.x;            gin[w][c0 + 1] = qv.y;
      gin[w][64 + c0] = nm0;        gin[w][64 + c0 + 1] = nm1;
      gin[w][128 + c0] = gv.x;      gin[w][128 + c0 + 1] = gv.y;
    }
    __syncthreads();
    if (act) {
      int c0 = 2 * lane;
#pragma unroll
      for (int u = 0; u < 2; ++u) {
        int c = c0 + u;
        float d = b1[c];
        for (int j = 0; j < 192; ++j) d += gin[w][j] * W1[j * 64 + c];
        float scl = bn1[c] * rsqrtf(bn1[192 + c] + EPSV);
        float val = (d - bn1[128 + c]) * scl + bn1[64 + c];
        hbuf[w][c] = fmaxf(val, 0.f);
      }
    }
    __syncthreads();
    if (act && lane < 8) {
      int g = lane;
      float d = b2[g];
#pragma unroll
      for (int c = 0; c < 64; ++c) d += hbuf[w][c] * W2[c * 8 + g];
      gate[(size_t)n * 8 + g] = 1.f / (1.f + __expf(-d));
    }
    __syncthreads();
  }
}

// ---------------------------------------------------------------------------
// Main kernel: one wave per point (16 neighbors = WMMA M dimension).
// All 15 weight fragments live in registers; features built as named scalars
// (no private arrays -> no scratch); LDS only for the two transposes + attn.
// ---------------------------------------------------------------------------
__global__ void main_kernel(const float* __restrict__ coord, const int* __restrict__ refidx,
                            const float* __restrict__ qws, const float* __restrict__ kws,
                            const float* __restrict__ vws, const float* __restrict__ gatews,
                            const float* __restrict__ peW1, const float* __restrict__ peb1,
                            const float* __restrict__ pebn, const float* __restrict__ peW2,
                            const float* __restrict__ peb2,
                            const float* __restrict__ weW1, const float* __restrict__ web1,
                            const float* __restrict__ webn, const float* __restrict__ weW2,
                            const float* __restrict__ web2,
                            float* __restrict__ out, int Npts, int tiles) {
  __shared__ _Float16 hstage[8][16 * 64];
  __shared__ _Float16 gstage[8][16 * 8];
  __shared__ float attst[8][16 * 8];

  int tid = threadIdx.x;
  int w = tid >> 5, lane = tid & 31, r = lane & 15, hi = lane >> 4;

  // Weight fragments in registers (120 VGPRs total).
  V16 w1f[4], w2f[4][2], u1f[2], u2f;
  float s1[4], t1[4], pb2[4];
#pragma unroll
  for (int t4 = 0; t4 < 4; ++t4) {
    int ch = r + 16 * t4;
    w1f[t4].v    = gathW(peW1, 64, ch, 0, hi, 28, 64);
    w2f[t4][0].v = gathW(peW2, 64, ch, 0, hi, 64, 64);
    w2f[t4][1].v = gathW(peW2, 64, ch, 32, hi, 64, 64);
    float s = pebn[ch] * rsqrtf(pebn[192 + ch] + EPSV);
    s1[t4] = s;
    t1[t4] = (peb1[ch] - pebn[128 + ch]) * s + pebn[64 + ch];
    pb2[t4] = peb2[ch];
  }
  u1f[0].v = gathW(weW1, 8, r, 0, hi, 64, 8);
  u1f[1].v = gathW(weW1, 8, r, 32, hi, 64, 8);
  u2f.v    = gathW(weW2, 8, r, 0, hi, 8, 8);
  float s2r = 0.f, t2r = 0.f, wb2 = 0.f;
  if (r < 8) {
    float s = webn[r] * rsqrtf(webn[24 + r] + EPSV);
    s2r = s;
    t2r = (web1[r] - webn[16 + r]) * s + webn[8 + r];
    wb2 = web2[r];
  }

  for (int t = blockIdx.x; t < tiles; t += gridDim.x) {
    int n = t * 8 + w;
    bool act = (n < Npts);
    V16 a20, a21, a30, a31;
    float maskf = 1.f;
    int my_i = 0;

    if (act) {
      int raw = refidx[n * 16 + r];
      maskf = (raw + 1 > 0) ? 1.f : 0.f;
      my_i = (raw < 0) ? 0 : raw;
      float px = coord[n * 3 + 0], py = coord[n * 3 + 1], pz = coord[n * 3 + 2];
      float dx = coord[my_i * 3 + 0] - px;
      float dy = coord[my_i * 3 + 1] - py;
      float dz = coord[my_i * 3 + 2] - pz;
      float dist = sqrtf(dx * dx + dy * dy + dz * dz);
      float s1x = __sinf(dx * PI_F),        s1y = __sinf(dy * PI_F);
      float c1x = __cosf(dx * PI_F),        c1y = __cosf(dy * PI_F);
      float s2x = __sinf(dx * 2.f * PI_F),  s2y = __sinf(dy * 2.f * PI_F);
      float c2x = __cosf(dx * 2.f * PI_F),  c2y = __cosf(dy * 2.f * PI_F);
      float s4x = __sinf(dx * 4.f * PI_F),  s4y = __sinf(dy * 4.f * PI_F);
      float c4x = __cosf(dx * 4.f * PI_F),  c4y = __cosf(dy * 4.f * PI_F);
      float s8x = __sinf(dx * 8.f * PI_F),  s8y = __sinf(dy * 8.f * PI_F);
      float c8x = __cosf(dx * 8.f * PI_F),  c8y = __cosf(dy * 8.f * PI_F);
      float s1z = __sinf(dz * PI_F),        c1z = __cosf(dz * PI_F);
      float s2z = __sinf(dz * 2.f * PI_F),  c2z = __cosf(dz * 2.f * PI_F);
      float s4z = __sinf(dz * 4.f * PI_F),  c4z = __cosf(dz * 4.f * PI_F);
      float s8z = __sinf(dz * 8.f * PI_F),  c8z = __cosf(dz * 8.f * PI_F);

      // A fragment for layer1: constant-index selects (K order per CDNA5 layout)
      V16 a1;
      a1.h[0]  = hsel(hi, dx,   s2x);   // K0  | K8
      a1.h[1]  = hsel(hi, dy,   s2y);   // K1  | K9
      a1.h[2]  = hsel(hi, dz,   c2x);   // K2  | K10
      a1.h[3]  = hsel(hi, dist, c2y);   // K3  | K11
      a1.h[4]  = hsel(hi, s1x,  s4x);   // K4  | K12
      a1.h[5]  = hsel(hi, s1y,  s4y);   // K5  | K13
      a1.h[6]  = hsel(hi, c1x,  c4x);   // K6  | K14
      a1.h[7]  = hsel(hi, c1y,  c4y);   // K7  | K15
      a1.h[8]  = hsel(hi, s8x,  s4z);   // K16 | K24
      a1.h[9]  = hsel(hi, s8y,  c4z);   // K17 | K25
      a1.h[10] = hsel(hi, c8x,  s8z);   // K18 | K26
      a1.h[11] = hsel(hi, c8y,  c8z);   // K19 | K27
      a1.h[12] = hsel(hi, s1z,  0.f);   // K20 | pad
      a1.h[13] = hsel(hi, c1z,  0.f);   // K21 | pad
      a1.h[14] = hsel(hi, s2z,  0.f);   // K22 | pad
      a1.h[15] = hsel(hi, c2z,  0.f);   // K23 | pad

      // Layer 1: xe(16x28) @ pe_W1 -> 16x64, bn+relu
#pragma unroll
      for (int t4 = 0; t4 < 4; ++t4) {
        int ch = r + 16 * t4;
        v8f acc = vzero8();
        acc = wmma16(a1.v, w1f[t4].v, acc);
#pragma unroll
        for (int i = 0; i < 8; ++i) {
          float hv = fmaxf(acc[i] * s1[t4] + t1[t4], 0.f);
          hstage[w][(i + 8 * hi) * 64 + ch] = (_Float16)hv;
        }
      }
    }
    __syncthreads();
    if (act) {  // re-read hidden in A layout (lane owns row r)
      const _Float16* hr = hstage[w] + r * 64;
#pragma unroll
      for (int e = 0; e < 8; ++e) {
        a20.h[e] = hr[8 * hi + e];      a20.h[8 + e] = hr[16 + 8 * hi + e];
        a21.h[e] = hr[32 + 8 * hi + e]; a21.h[8 + e] = hr[48 + 8 * hi + e];
      }
    }
    __syncthreads();
    if (act) {  // Layer 2: -> rpe 16x64 (+bias), back to LDS
#pragma unroll
      for (int t4 = 0; t4 < 4; ++t4) {
        int ch = r + 16 * t4;
        v8f acc = vzero8();
        acc = wmma16(a20.v, w2f[t4][0].v, acc);
        acc = wmma16(a21.v, w2f[t4][1].v, acc);
#pragma unroll
        for (int i = 0; i < 8; ++i)
          hstage[w][(i + 8 * hi) * 64 + ch] = (_Float16)(acc[i] + pb2[t4]);
      }
    }
    __syncthreads();
    if (act) {  // relation = k[idx] - q[n] + rpe, in A layout; -> 16x8 bn,relu
      const _Float16* hr = hstage[w] + r * 64;
      const float* krow = kws + (size_t)my_i * 64;
      const float* qrow = qws + (size_t)n * 64;
#pragma unroll
      for (int e = 0; e < 8; ++e) {
        int k0 = 8 * hi + e, k1 = 16 + 8 * hi + e;
        a30.h[e]     = (_Float16)((float)hr[k0] + krow[k0] - qrow[k0]);
        a30.h[8 + e] = (_Float16)((float)hr[k1] + krow[k1] - qrow[k1]);
        a31.h[e]     = (_Float16)((float)hr[32 + k0] + krow[32 + k0] - qrow[32 + k0]);
        a31.h[8 + e] = (_Float16)((float)hr[32 + k1] + krow[32 + k1] - qrow[32 + k1]);
      }
      v8f acc3 = vzero8();
      acc3 = wmma16(a30.v, u1f[0].v, acc3);
      acc3 = wmma16(a31.v, u1f[1].v, acc3);
      if (r < 8) {
#pragma unroll
        for (int i = 0; i < 8; ++i) {
          float gv = fmaxf(acc3[i] * s2r + t2r, 0.f);
          gstage[w][(i + 8 * hi) * 8 + r] = (_Float16)gv;
        }
      }
    }
    __syncthreads();
    if (act) {  // 16x8 @ we_W2 -> geom, masked softmax over the 16 neighbors
      V16 a4;
#pragma unroll
      for (int e = 0; e < 8; ++e) {
        a4.h[e]     = (hi == 0) ? gstage[w][r * 8 + e] : (_Float16)0.f;
        a4.h[8 + e] = (_Float16)0.f;
      }
      v8f acc4 = vzero8();
      acc4 = wmma16(a4.v, u2f.v, acc4);

      float gt = (r < 8) ? gatews[(size_t)n * 8 + r] : 0.f;
      float zrow[8], ex[8];
      float mx = -1e30f;
#pragma unroll
      for (int i = 0; i < 8; ++i) {
        zrow[i] = (acc4[i] + wb2) * gt;
        mx = fmaxf(mx, zrow[i]);
      }
      mx = fmaxf(mx, __shfl_xor(mx, 16));
      float sum = 0.f;
#pragma unroll
      for (int i = 0; i < 8; ++i) { ex[i] = __expf(zrow[i] - mx); sum += ex[i]; }
      sum += __shfl_xor(sum, 16);
      float inv = 1.f / sum;
#pragma unroll
      for (int i = 0; i < 8; ++i) {
        float mm = __shfl(maskf, i + 8 * hi);
        if (r < 8) attst[w][(i + 8 * hi) * 8 + r] = ex[i] * inv * mm;
      }
    }
    __syncthreads();
    if (act) {  // out[n][c] = sum_m attn[m][c>>3] * v[idx[m]][c]
      int c0 = 2 * lane, g = c0 >> 3;
      float o0 = 0.f, o1 = 0.f;
#pragma unroll
      for (int m = 0; m < 16; ++m) {
        int im = __shfl(my_i, m);            // lane m holds idx[m]
        float a = attst[w][m * 8 + g];
        float2 vv = *(const float2*)(vws + (size_t)im * 64 + c0);
        o0 += a * vv.x; o1 += a * vv.y;
      }
      float2 res; res.x = o0; res.y = o1;
      *(float2*)(out + (size_t)n * 64 + c0) = res;
    }
    __syncthreads();
  }
}

// ---------------------------------------------------------------------------
extern "C" void kernel_launch(void* const* d_in, const int* in_sizes, int n_in,
                              void* d_out, int out_size, void* d_ws, size_t ws_size,
                              hipStream_t stream) {
  (void)n_in; (void)out_size; (void)ws_size;
  const float* feat  = (const float*)d_in[0];
  const float* coord = (const float*)d_in[1];
  const int*   ridx  = (const int*)d_in[2];
  const int*   offs  = (const int*)d_in[3];
  const float* Wq = (const float*)d_in[4];
  const float* bq = (const float*)d_in[5];
  const float* bnq = (const float*)d_in[6];
  const float* Wk = (const float*)d_in[7];
  const float* bk = (const float*)d_in[8];
  const float* bnk = (const float*)d_in[9];
  const float* Wv = (const float*)d_in[10];
  const float* bv = (const float*)d_in[11];
  const float* peW1 = (const float*)d_in[12];
  const float* peb1 = (const float*)d_in[13];
  const float* pebn = (const float*)d_in[14];
  const float* peW2 = (const float*)d_in[15];
  const float* peb2 = (const float*)d_in[16];
  const float* weW1 = (const float*)d_in[17];
  const float* web1 = (const float*)d_in[18];
  const float* webn = (const float*)d_in[19];
  const float* weW2 = (const float*)d_in[20];
  const float* web2 = (const float*)d_in[21];
  const float* cgW1 = (const float*)d_in[22];
  const float* cgb1 = (const float*)d_in[23];
  const float* cgbn = (const float*)d_in[24];
  const float* cgW2 = (const float*)d_in[25];
  const float* cgb2 = (const float*)d_in[26];

  int Npts = in_sizes[0] / 64;
  float* ws   = (float*)d_ws;
  float* qws  = ws;
  float* kws  = qws + (size_t)Npts * 64;
  float* vws  = kws + (size_t)Npts * 64;
  float* segm = vws + (size_t)Npts * 64;
  float* gate = segm + 256;
  float* out  = (float*)d_out;

  int tiles16 = (Npts + 15) / 16;
  int blocksA = (tiles16 + 7) / 8; if (blocksA > 2048) blocksA = 2048;
  proj_kernel<<<blocksA, 256, 0, stream>>>(feat, Wq, bq, bnq, qws, Npts, 1);
  proj_kernel<<<blocksA, 256, 0, stream>>>(feat, Wk, bk, bnk, kws, Npts, 1);
  proj_kernel<<<blocksA, 256, 0, stream>>>(feat, Wv, bv, bnq /*unused*/, vws, Npts, 0);

  seg_init<<<1, 256, 0, stream>>>(segm);
  seg_max<<<1024, 256, 0, stream>>>(qws, offs, (unsigned int*)segm, Npts);

  int tiles8 = (Npts + 7) / 8;
  int blocksG = tiles8 < 2048 ? tiles8 : 2048;
  gate_kernel<<<blocksG, 256, 0, stream>>>(qws, segm, ridx, offs, cgW1, cgb1, cgbn,
                                           cgW2, cgb2, gate, Npts, tiles8);

  main_kernel<<<blocksG, 256, 0, stream>>>(coord, ridx, qws, kws, vws, gate,
                                           peW1, peb1, pebn, peW2, peb2,
                                           weW1, web1, webn, weW2, web2,
                                           out, Npts, tiles8);
}
